// _BiMambaBlock_56358560858364
// MI455X (gfx1250) — matrix operand order repaired and checked by reading, not compiled, not run on hardware
//
#include <hip/hip_runtime.h>
#include <math.h>

#define DIM    512
#define DSTATE 16
#define DCONV  4
#define DINNER 1024
#define DTRANK 32
#define NBATCH 4
#define SEQ    2048
#define MROWS  (NBATCH * SEQ)   // 8192 rows, multiple of 32

typedef __attribute__((ext_vector_type(16))) _Float16 v16h;
typedef __attribute__((ext_vector_type(8)))  _Float16 v8h;
typedef __attribute__((ext_vector_type(8)))  float    v8f;

// ---------------------------------------------------------------------------
// LayerNorm over DIM=512, output f16 row (WMMA A operand precision).
// One block per row, 256 threads x 2 elements.
// ---------------------------------------------------------------------------
__global__ __launch_bounds__(256) void layernorm_f16(
    const float* __restrict__ x, const float* __restrict__ g,
    const float* __restrict__ b, _Float16* __restrict__ wh)
{
    __shared__ float red[256];
    const int row = blockIdx.x;
    const int t   = threadIdx.x;
    const float* xr = x + (size_t)row * DIM;
    float v0 = xr[t], v1 = xr[t + 256];
    red[t] = v0 + v1;
    __syncthreads();
    for (int s = 128; s > 0; s >>= 1) { if (t < s) red[t] += red[t + s]; __syncthreads(); }
    const float mean = red[0] * (1.0f / DIM);
    __syncthreads();
    const float d0 = v0 - mean, d1 = v1 - mean;
    red[t] = d0 * d0 + d1 * d1;
    __syncthreads();
    for (int s = 128; s > 0; s >>= 1) { if (t < s) red[t] += red[t + s]; __syncthreads(); }
    const float rstd = rsqrtf(red[0] * (1.0f / DIM) + 1e-5f);
    _Float16* wr = wh + (size_t)row * DIM;
    wr[t]       = (_Float16)(d0 * rstd * g[t]       + b[t]);
    wr[t + 256] = (_Float16)(d1 * rstd * g[t + 256] + b[t + 256]);
}

// ---------------------------------------------------------------------------
// Pack a row-major f32 weight (K x N) into WMMA B-fragment layout, f16:
//   Bp[((kt*tiles_n + nt)*32 + lane)*16 + j]
//   lane 0-15 : col = nt*16+lane,    K = kt*32 + j      (j = 0..15)
//   lane 16-31: col = nt*16+lane-16, K = kt*32 + 16 + j
// (contiguous K order k = 16h + j; the GEMM's A loads use the same order)
// ---------------------------------------------------------------------------
__global__ void pack_b_f16(const float* __restrict__ W, _Float16* __restrict__ Bp,
                           int K, int N)
{
    size_t i = (size_t)blockIdx.x * blockDim.x + threadIdx.x;
    if (i >= (size_t)K * N) return;
    const int j       = (int)(i & 15);
    const int lane    = (int)((i >> 4) & 31);
    const size_t tile = i >> 9;
    const int tiles_n = N >> 4;
    const int kt  = (int)(tile / tiles_n);
    const int nt  = (int)(tile - (size_t)kt * tiles_n);
    const int col = (nt << 4) + (lane & 15);
    const int kr  = (kt << 5) + ((lane < 16) ? j : 16 + j);
    Bp[i] = (_Float16)W[(size_t)kr * N + col];
}

// ---------------------------------------------------------------------------
// WMMA GEMM: C(MROWS x N, f32) = A(MROWS x K, f16 row-major) * Bp(packed f16)
// One wave computes a 32x64 strip: 8 accumulators; each K-step reuses 2 A
// fragments across 4 B fragments (8 WMMAs per 6 fragment loads).
// K multiple of 32, N multiple of 64, MROWS multiple of 32.
// ---------------------------------------------------------------------------
__device__ __forceinline__ void store_tile(float* __restrict__ cp, v8f a,
                                           int N, int accumulate)
{
    union { v8f v; float f[8]; } u; u.v = a;
    if (accumulate) {
#pragma unroll
        for (int i = 0; i < 8; ++i) cp[(size_t)i * N] += u.f[i];
    } else {
#pragma unroll
        for (int i = 0; i < 8; ++i) cp[(size_t)i * N] = u.f[i];
    }
}

#define WMMA_F16(A_, B_, C_) \
    __builtin_amdgcn_wmma_f32_16x16x32_f16(false, (A_), false, (B_), (short)0, (C_), false, false)

__global__ __launch_bounds__(256) void gemm_f16_wmma(
    const _Float16* __restrict__ A, const _Float16* __restrict__ Bp,
    float* __restrict__ C, int K, int N, int accumulate)
{
    const int lane = threadIdx.x & 31;
    const int wave = blockIdx.x * (blockDim.x >> 5) + (threadIdx.x >> 5);
    const int ngrp = N >> 6;                  // groups of 4 n-tiles
    const int tm   = wave / ngrp;             // 32-row M tile
    const int tg   = wave - tm * ngrp;
    if (tm >= (MROWS >> 5)) return;           // wave-uniform: EXEC stays all-1s

    const int row0 = (tm << 5) + (lane & 15);
    const int kb   = (lane < 16) ? 0 : 8;     // A: K sub-base; C: M sub-base
    const int tiles_n = N >> 4;

    v8f c00 = {}, c01 = {}, c02 = {}, c03 = {};
    v8f c10 = {}, c11 = {}, c12 = {}, c13 = {};
    const _Float16* __restrict__ arow0 = A + (size_t)row0 * K;
    const _Float16* __restrict__ arow1 = arow0 + (size_t)16 * K;
    for (int k0 = 0; k0 < K; k0 += 32) {
        union { v16h v; v8h h[2]; } a0, a1;   // contiguous K: half h holds K 16h..16h+15
        a0.h[0] = *(const v8h*)(arow0 + k0 + 2 * kb);
        a0.h[1] = *(const v8h*)(arow0 + k0 + 2 * kb + 8);
        a1.h[0] = *(const v8h*)(arow1 + k0 + 2 * kb);
        a1.h[1] = *(const v8h*)(arow1 + k0 + 2 * kb + 8);
        const _Float16* __restrict__ bb =
            Bp + (((size_t)(k0 >> 5) * tiles_n + ((size_t)tg << 2)) << 9) + (lane << 4);
        v16h b0 = *(const v16h*)(bb);
        v16h b1 = *(const v16h*)(bb + 512);
        v16h b2 = *(const v16h*)(bb + 1024);
        v16h b3 = *(const v16h*)(bb + 1536);
        c00 = WMMA_F16(a0.v, b0, c00);
        c10 = WMMA_F16(a1.v, b0, c10);
        c01 = WMMA_F16(a0.v, b1, c01);
        c11 = WMMA_F16(a1.v, b1, c11);
        c02 = WMMA_F16(a0.v, b2, c02);
        c12 = WMMA_F16(a1.v, b2, c12);
        c03 = WMMA_F16(a0.v, b3, c03);
        c13 = WMMA_F16(a1.v, b3, c13);
    }
    float* cb0 = C + (size_t)((tm << 5) + kb) * N + (tg << 6) + (lane & 15);
    float* cb1 = cb0 + (size_t)16 * N;
    store_tile(cb0,      c00, N, accumulate);
    store_tile(cb0 + 16, c01, N, accumulate);
    store_tile(cb0 + 32, c02, N, accumulate);
    store_tile(cb0 + 48, c03, N, accumulate);
    store_tile(cb1,      c10, N, accumulate);
    store_tile(cb1 + 16, c11, N, accumulate);
    store_tile(cb1 + 32, c12, N, accumulate);
    store_tile(cb1 + 48, c13, N, accumulate);
}

// ---------------------------------------------------------------------------
// Depthwise causal conv (k=4) + SiLU. dir=+1: taps t-3..t; dir=-1: taps t..t+3
// (backward direction computed in natural time order, flips cancel).
// ---------------------------------------------------------------------------
__global__ void conv_silu(const float* __restrict__ xz, const float* __restrict__ cw,
                          const float* __restrict__ cb, float* __restrict__ xc,
                          _Float16* __restrict__ xch, int dir)
{
    size_t idx = (size_t)blockIdx.x * blockDim.x + threadIdx.x;
    if (idx >= (size_t)MROWS * DINNER) return;
    const int    d  = (int)(idx & (DINNER - 1));
    const size_t rt = idx >> 10;
    const int    t  = (int)(rt & (SEQ - 1));
    const int    bi = (int)(rt >> 11);
    float s = cb[d];
#pragma unroll
    for (int k = 0; k < DCONV; ++k) {
        const int tt = (dir > 0) ? (t + k - (DCONV - 1)) : (t + (DCONV - 1) - k);
        if (tt >= 0 && tt < SEQ)
            s += cw[k * DINNER + d] * xz[(((size_t)bi * SEQ + tt) << 11) + d];
    }
    const float sil = s / (1.0f + __expf(-s));
    xc[idx]  = sil;
    xch[idx] = (_Float16)sil;
}

// Extract first DTRANK=32 cols of dbc (MROWS x 64 f32) as f16 A operand.
__global__ void extract_dt_in(const float* __restrict__ dbc, _Float16* __restrict__ dbch)
{
    size_t idx = (size_t)blockIdx.x * blockDim.x + threadIdx.x;
    if (idx >= (size_t)MROWS * DTRANK) return;
    const int c = (int)(idx & 31);
    const size_t row = idx >> 5;
    dbch[idx] = (_Float16)dbc[(row << 6) + c];
}

// dt = softplus(dt_raw + dt_b)
__global__ void dt_softplus(float* __restrict__ dt, const float* __restrict__ dtb)
{
    size_t idx = (size_t)blockIdx.x * blockDim.x + threadIdx.x;
    if (idx >= (size_t)MROWS * DINNER) return;
    const int d = (int)(idx & (DINNER - 1));
    const float v = dt[idx] + dtb[d];
    dt[idx] = (v > 20.0f) ? v : log1pf(__expf(v));
}

// ---------------------------------------------------------------------------
// Selective scan. One thread per (batch, channel): 16-state recurrence in
// registers, sequential over SEQ (reversed for dir=-1).
// ---------------------------------------------------------------------------
__global__ __launch_bounds__(256) void selective_scan(
    const float* __restrict__ dt, const float* __restrict__ dbc,
    const float* __restrict__ xc, const float* __restrict__ A_log,
    float* __restrict__ ys, int dir)
{
    const int idx = blockIdx.x * blockDim.x + threadIdx.x;
    if (idx >= NBATCH * DINNER) return;
    const int bi = idx >> 10;
    const int d  = idx & (DINNER - 1);
    float Av[DSTATE], h[DSTATE];
#pragma unroll
    for (int s = 0; s < DSTATE; ++s) { Av[s] = -__expf(A_log[d * DSTATE + s]); h[s] = 0.0f; }
    for (int step = 0; step < SEQ; ++step) {
        const int t = (dir > 0) ? step : (SEQ - 1 - step);
        const size_t row = (size_t)bi * SEQ + t;
        const float dtv = dt[(row << 10) + d];
        const float xv  = xc[(row << 10) + d];
        union { float4 q[8]; float f[32]; } bc;   // B then C (16 each)
        const float4* bp = (const float4*)(dbc + (row << 6) + DTRANK);
#pragma unroll
        for (int i = 0; i < 8; ++i) bc.q[i] = bp[i];
        const float dx = dtv * xv;
        float y = 0.0f;
#pragma unroll
        for (int s = 0; s < DSTATE; ++s) {
            h[s] = __expf(dtv * Av[s]) * h[s] + dx * bc.f[s];
            y += h[s] * bc.f[16 + s];
        }
        ys[(row << 10) + d] = y;
    }
}

// y = (ys + xc*Dp) * silu(z), cast f16 for the output GEMM.
__global__ void y_epilogue(const float* __restrict__ ys, const float* __restrict__ xc,
                           const float* __restrict__ xz, const float* __restrict__ Dp,
                           _Float16* __restrict__ yh)
{
    size_t idx = (size_t)blockIdx.x * blockDim.x + threadIdx.x;
    if (idx >= (size_t)MROWS * DINNER) return;
    const int d = (int)(idx & (DINNER - 1));
    const size_t row = idx >> 10;
    const float z = xz[(row << 11) + DINNER + d];
    const float y = (ys[idx] + xc[idx] * Dp[d]) * (z / (1.0f + __expf(-z)));
    yh[idx] = (_Float16)y;
}

// out = x + 0.5 * (acc_fwd + acc_bwd)   (acc already holds the sum)
__global__ void final_out(const float* __restrict__ x, const float* __restrict__ acc,
                          float* __restrict__ out)
{
    size_t idx = (size_t)blockIdx.x * blockDim.x + threadIdx.x;
    if (idx >= (size_t)MROWS * DIM) return;
    out[idx] = x[idx] + 0.5f * acc[idx];
}

// ---------------------------------------------------------------------------
// Host-side pipeline
// ---------------------------------------------------------------------------
static inline int ceil_div_i(size_t a, size_t b) { return (int)((a + b - 1) / b); }

extern "C" void kernel_launch(void* const* d_in, const int* in_sizes, int n_in,
                              void* d_out, int out_size, void* d_ws, size_t ws_size,
                              hipStream_t stream)
{
    (void)in_sizes; (void)n_in; (void)out_size; (void)ws_size;
    const float* x    = (const float*)d_in[0];
    const float* ln_g = (const float*)d_in[1];
    const float* ln_b = (const float*)d_in[2];
    // per-direction params: in_w, conv_w, conv_b, xproj_w, dt_w, dt_b, A_log, Dp, out_w
    auto PRM = [&](int dirIdx, int which) -> const float* {
        return (const float*)d_in[3 + dirIdx * 9 + which];
    };

    char* ws = (char*)d_ws;
    size_t off = 0;
    auto alloc = [&](size_t bytes) -> char* {
        char* p = ws + off; off += (bytes + 255) & ~(size_t)255; return p;
    };
    _Float16* wh   = (_Float16*)alloc((size_t)MROWS * DIM * 2);
    float*    accb = (float*)   alloc((size_t)MROWS * DIM * 4);
    float*    xz   = (float*)   alloc((size_t)MROWS * 2 * DINNER * 4);
    float*    xc   = (float*)   alloc((size_t)MROWS * DINNER * 4);
    _Float16* xch  = (_Float16*)alloc((size_t)MROWS * DINNER * 2);
    float*    dbc  = (float*)   alloc((size_t)MROWS * 64 * 4);
    _Float16* dbch = (_Float16*)alloc((size_t)MROWS * DTRANK * 2);
    float*    dtb  = (float*)   alloc((size_t)MROWS * DINNER * 4);
    float*    ysb  = (float*)   alloc((size_t)MROWS * DINNER * 4);
    _Float16* yh   = (_Float16*)alloc((size_t)MROWS * DINNER * 2);
    _Float16* inwp = (_Float16*)alloc((size_t)DIM * 2 * DINNER * 2);
    _Float16* xpjp = (_Float16*)alloc((size_t)DINNER * 64 * 2);
    _Float16* dtwp = (_Float16*)alloc((size_t)DTRANK * DINNER * 2);
    _Float16* outp = (_Float16*)alloc((size_t)DINNER * DIM * 2);

    auto gemm = [&](const _Float16* A, const _Float16* Bp, float* C,
                    int K, int N, int accum) {
        const int waves  = (MROWS / 32) * (N / 64);
        const int blocks = (waves + 7) / 8;   // 8 waves / 256-thread block
        gemm_f16_wmma<<<blocks, 256, 0, stream>>>(A, Bp, C, K, N, accum);
    };
    auto pack = [&](const float* W, _Float16* Bp, int K, int N) {
        pack_b_f16<<<ceil_div_i((size_t)K * N, 256), 256, 0, stream>>>(W, Bp, K, N);
    };

    layernorm_f16<<<MROWS, 256, 0, stream>>>(x, ln_g, ln_b, wh);

    const size_t nMD = (size_t)MROWS * DINNER;
    for (int dirIdx = 0; dirIdx < 2; ++dirIdx) {
        const int dir = (dirIdx == 0) ? 1 : -1;
        pack(PRM(dirIdx, 0), inwp, DIM, 2 * DINNER);
        pack(PRM(dirIdx, 3), xpjp, DINNER, DTRANK + 2 * DSTATE);
        pack(PRM(dirIdx, 4), dtwp, DTRANK, DINNER);
        pack(PRM(dirIdx, 8), outp, DINNER, DIM);

        // xz = w @ in_w   (8192x512 * 512x2048)
        gemm(wh, inwp, xz, DIM, 2 * DINNER, 0);
        // depthwise conv + silu
        conv_silu<<<ceil_div_i(nMD, 256), 256, 0, stream>>>(
            xz, PRM(dirIdx, 1), PRM(dirIdx, 2), xc, xch, dir);
        // dbc = xc @ xproj_w  (K=1024, N=64)
        gemm(xch, xpjp, dbc, DINNER, 64, 0);
        // dt = softplus(dbc[:, :32] @ dt_w + dt_b)
        extract_dt_in<<<ceil_div_i((size_t)MROWS * DTRANK, 256), 256, 0, stream>>>(dbc, dbch);
        gemm(dbch, dtwp, dtb, DTRANK, DINNER, 0);
        dt_softplus<<<ceil_div_i(nMD, 256), 256, 0, stream>>>(dtb, PRM(dirIdx, 5));
        // sequential selective scan
        selective_scan<<<(NBATCH * DINNER) / 256, 256, 0, stream>>>(
            dtb, dbc, xc, PRM(dirIdx, 6), ysb, dir);
        // gated epilogue -> f16
        y_epilogue<<<ceil_div_i(nMD, 256), 256, 0, stream>>>(ysb, xc, xz, PRM(dirIdx, 7), yh);
        // acc (+)= y @ out_w  (K=1024, N=512); fwd writes, bwd accumulates
        gemm(yh, outp, accb, DINNER, DIM, dirIdx);
    }
    final_out<<<ceil_div_i((size_t)MROWS * DIM, 256), 256, 0, stream>>>(
        x, accb, (float*)d_out);
}
